// TinyCausalWanModel_75471165325704
// MI455X (gfx1250) — compile-verified, matrix operand order checked by reading, not run on hardware
//
#include <hip/hip_runtime.h>

#define L_TOK 3840
#define DIM   1536
#define NH    12
#define HD    128
#define FRAME 480
#define EPS_F 1e-6f

typedef __attribute__((ext_vector_type(16))) __bf16 v16bf;
typedef __attribute__((ext_vector_type(8)))  float  v8f;
typedef __attribute__((ext_vector_type(4)))  unsigned int u32x4;
typedef __attribute__((ext_vector_type(4)))  int i32x4;
typedef __attribute__((ext_vector_type(8)))  int i32x8;

union Frag16 {
    uint4 u[2];
    v16bf v;
};

__device__ __forceinline__ unsigned short f2bf(float f) {
    unsigned int u = __builtin_bit_cast(unsigned int, f);
    u += 0x7FFFu + ((u >> 16) & 1u);
    return (unsigned short)(u >> 16);
}
__device__ __forceinline__ float bf2f(unsigned short h) {
    unsigned int u = ((unsigned int)h) << 16;
    return __builtin_bit_cast(float, u);
}
__device__ __forceinline__ v8f vzero() {
    v8f v;
#pragma unroll
    for (int i = 0; i < 8; i++) v[i] = 0.f;
    return v;
}
__device__ __forceinline__ v8f wmma_bf16(v16bf a, v16bf b, v8f c) {
    // D = A(16x32 bf16) * B(32x16 bf16) + C(16x16 f32)
    return __builtin_amdgcn_wmma_f32_16x16x32_bf16(false, a, false, b,
                                                   (short)0, c, false, false);
}

// ---------------------------------------------------------------------------
// TDM: issue a 2D bf16 tile load (tile_d1 rows x tile_d0 elems, row stride
// stride_d0 elems) from global memory into LDS at byte offset lds_off.
// Descriptor bitfields per CDNA5 ISA 8.3/8.4 (D# group0 128b, group1 256b).
// Wave-level DMA: tracked with TENSORcnt, EXEC ignored.
// ---------------------------------------------------------------------------
__device__ __forceinline__ void tdm_load_2d(
    unsigned lds_off, const void* gaddr,
    unsigned tensor_d0, unsigned tensor_d1,
    unsigned tile_d0, unsigned tile_d1, unsigned stride_d0)
{
    unsigned long long ga = (unsigned long long)(size_t)gaddr;
    u32x4 g0;
    g0[0] = 1u;                                              // count=1, user D#
    g0[1] = lds_off;                                         // lds_addr (bytes)
    g0[2] = (unsigned)(ga & 0xFFFFFFFFu);                    // global_addr[31:0]
    g0[3] = (unsigned)((ga >> 32) & 0x01FFFFFFu) | (2u << 30); // [56:32] + type=2
    i32x8 g1;
    g1[0] = (int)(1u << 16);                                 // data_size=1 (2B)
    g1[1] = (int)((tensor_d0 & 0xFFFFu) << 16);              // tensor_dim0[15:0]
    g1[2] = (int)((tensor_d0 >> 16) | ((tensor_d1 & 0xFFFFu) << 16));
    g1[3] = (int)((tensor_d1 >> 16) | ((tile_d0 & 0xFFFFu) << 16));
    g1[4] = (int)(tile_d1 & 0xFFFFu);                        // tile_dim1, dim2=0
    g1[5] = (int)stride_d0;                                  // dim0_stride[31:0]
    g1[6] = 0;                                               // stride hi, dim1_stride (2D)
    g1[7] = 0;
    i32x4 z4 = {0, 0, 0, 0};
#if defined(__clang_major__) && (__clang_major__ >= 23)
    i32x8 z8 = {0, 0, 0, 0, 0, 0, 0, 0};
    __builtin_amdgcn_tensor_load_to_lds(g0, g1, z4, z4, z8, 0);
#else
    __builtin_amdgcn_tensor_load_to_lds(g0, g1, z4, z4, 0);
#endif
}

// ---------------------------------------------------------------------------
// Cast fp32 -> bf16, vectorized x4
// ---------------------------------------------------------------------------
__global__ __launch_bounds__(256) void cast_f32_bf16(
    const float* __restrict__ in, unsigned short* __restrict__ out, int n)
{
    int i = (blockIdx.x * 256 + threadIdx.x) * 4;
    if (i + 3 < n) {
        float4 v = *(const float4*)(in + i);
        unsigned int lo = (unsigned int)f2bf(v.x) | ((unsigned int)f2bf(v.y) << 16);
        unsigned int hi = (unsigned int)f2bf(v.z) | ((unsigned int)f2bf(v.w) << 16);
        uint2 pk; pk.x = lo; pk.y = hi;
        *(uint2*)(out + i) = pk;
    }
}

// ---------------------------------------------------------------------------
// Cast + transpose weight: Wt[n][k] = (bf16)W[k][n]   (dim x dim)
// ---------------------------------------------------------------------------
__global__ __launch_bounds__(256) void cast_transpose(
    const float* __restrict__ W, unsigned short* __restrict__ Wt, int dim)
{
    __shared__ float tile[32][33];
    const int bx = blockIdx.x * 32;   // n-range
    const int by = blockIdx.y * 32;   // k-range
    const int t = threadIdx.x;
#pragma unroll
    for (int i = t; i < 1024; i += 256) {
        int r = i >> 5, c = i & 31;
        tile[r][c] = W[(size_t)(by + r) * dim + bx + c];
    }
    __syncthreads();
#pragma unroll
    for (int i = t; i < 1024; i += 256) {
        int r = i >> 5, c = i & 31;
        Wt[(size_t)(bx + r) * dim + by + c] = f2bf(tile[c][r]);
    }
}

// ---------------------------------------------------------------------------
// WMMA GEMM: C[M][N] = A[M][K](bf16) * Bt[N][K]^T(bf16) + bias[N]
// Block: 256 thr (8 waves), tile 128x128, K-step 32.
// TDM double-buffered staging: wave 0 issues tensor_load_to_lds for the next
// K-slab while all waves run WMMAs on the current one; s_wait_tensorcnt gates
// the buffer swap.
// ---------------------------------------------------------------------------
template <bool OUT_BF16>
__global__ __launch_bounds__(256) void gemm_bf16(
    const unsigned short* __restrict__ A,
    const unsigned short* __restrict__ Bt,
    const float* __restrict__ bias,
    void* __restrict__ Cp,
    int M, int N, int K)
{
    __shared__ unsigned short sA[2][128 * 32];
    __shared__ unsigned short sB[2][128 * 32];

    const int t     = threadIdx.x;
    const int lane  = t & 31;
    const int wave  = t >> 5;
    const int tileN = blockIdx.x * 128;
    const int tileM = blockIdx.y * 128;
    const int m0    = (wave & 3) * 32;
    const int n0    = (wave >> 2) * 64;
    const int lr16  = lane & 15;
    const int hl    = lane >> 4;      // 0 | 1
    const int kb    = hl * 8;         // A/B K-base per ISA lane layout

    const unsigned ldsA[2] = { (unsigned)(size_t)&sA[0][0], (unsigned)(size_t)&sA[1][0] };
    const unsigned ldsB[2] = { (unsigned)(size_t)&sB[0][0], (unsigned)(size_t)&sB[1][0] };

    v8f acc[2][4];
#pragma unroll
    for (int mi = 0; mi < 2; mi++)
#pragma unroll
        for (int ni = 0; ni < 4; ni++) acc[mi][ni] = vzero();

    auto issue = [&](int buf, int k0) {
        tdm_load_2d(ldsA[buf], A + (size_t)tileM * K + k0,
                    (unsigned)K, (unsigned)M, 32u, 128u, (unsigned)K);
        tdm_load_2d(ldsB[buf], Bt + (size_t)tileN * K + k0,
                    (unsigned)K, (unsigned)N, 32u, 128u, (unsigned)K);
    };

    const int nk = K / 32;
    if (wave == 0) {
        issue(0, 0);
        __builtin_amdgcn_s_wait_tensorcnt(0);
    }
    __syncthreads();

    for (int i = 0; i < nk; i++) {
        const int cur = i & 1;
        if (wave == 0 && i + 1 < nk) issue(cur ^ 1, (i + 1) * 32);

        Frag16 fa[2], fb[4];
#pragma unroll
        for (int mi = 0; mi < 2; mi++) {
            const unsigned short* p = &sA[cur][(m0 + 16 * mi + lr16) * 32 + kb];
            fa[mi].u[0] = *(const uint4*)p;
            fa[mi].u[1] = *(const uint4*)(p + 16);
        }
#pragma unroll
        for (int ni = 0; ni < 4; ni++) {
            const unsigned short* p = &sB[cur][(n0 + 16 * ni + lr16) * 32 + kb];
            fb[ni].u[0] = *(const uint4*)p;
            fb[ni].u[1] = *(const uint4*)(p + 16);
        }
#pragma unroll
        for (int mi = 0; mi < 2; mi++)
#pragma unroll
            for (int ni = 0; ni < 4; ni++)
                acc[mi][ni] = wmma_bf16(fa[mi].v, fb[ni].v, acc[mi][ni]);

        __syncthreads();                  // everyone done reading sA/sB[cur]
        if (wave == 0 && i + 1 < nk)
            __builtin_amdgcn_s_wait_tensorcnt(0); // next buffer landed
        __syncthreads();                  // release compute on next buffer
    }

    // Epilogue: C layout -> lane = col (0..15), reg r -> row r + 8*hl
#pragma unroll
    for (int ni = 0; ni < 4; ni++) {
        const int col = tileN + n0 + 16 * ni + lr16;
        const float bv = bias ? bias[col] : 0.f;
#pragma unroll
        for (int mi = 0; mi < 2; mi++) {
#pragma unroll
            for (int r = 0; r < 8; r++) {
                const int row = tileM + m0 + 16 * mi + r + hl * 8;
                const float vv = acc[mi][ni][r] + bv;
                if (OUT_BF16)
                    ((unsigned short*)Cp)[(size_t)row * N + col] = f2bf(vv);
                else
                    ((float*)Cp)[(size_t)row * N + col] = vv;
            }
        }
    }
}

// ---------------------------------------------------------------------------
// RMSNorm (over full DIM) + 3D RoPE + layout shuffle.
// In:  Qg/Kg/Vg  bf16 [L][DIM]   (x@W + b)
// Out: qb/kb     bf16 [NH][L][HD]  (normed + roped)
//      vT        bf16 [NH][HD][L]  (transposed for PV B-fragments)
// ---------------------------------------------------------------------------
__global__ __launch_bounds__(256) void norm_rope_kernel(
    const unsigned short* __restrict__ Qg,
    const unsigned short* __restrict__ Kg,
    const unsigned short* __restrict__ Vg,
    const float* __restrict__ gq, const float* __restrict__ gk,
    const float* __restrict__ ff, const float* __restrict__ fh,
    const float* __restrict__ fw,
    unsigned short* __restrict__ qb, unsigned short* __restrict__ kb,
    unsigned short* __restrict__ vT)
{
    const int pos  = blockIdx.x;
    const int t    = threadIdx.x;
    const int lane = t & 31, wave = t >> 5;
    const unsigned short* qrow = Qg + (size_t)pos * DIM;
    const unsigned short* krow = Kg + (size_t)pos * DIM;
    const unsigned short* vrow = Vg + (size_t)pos * DIM;

    float sq = 0.f, sk = 0.f;
    for (int i = t; i < DIM; i += 256) {
        float q = bf2f(qrow[i]);
        float k = bf2f(krow[i]);
        sq += q * q; sk += k * k;
    }
#pragma unroll
    for (int o = 16; o >= 1; o >>= 1) {
        sq += __shfl_xor(sq, o, 32);
        sk += __shfl_xor(sk, o, 32);
    }
    __shared__ float red[2][8];
    __shared__ float rs[2];
    if (lane == 0) { red[0][wave] = sq; red[1][wave] = sk; }
    __syncthreads();
    if (t == 0) {
        float a = 0.f, b = 0.f;
#pragma unroll
        for (int w = 0; w < 8; w++) { a += red[0][w]; b += red[1][w]; }
        rs[0] = rsqrtf(a / (float)DIM + EPS_F);
        rs[1] = rsqrtf(b / (float)DIM + EPS_F);
    }
    __syncthreads();
    const float rq = rs[0], rk = rs[1];

    const int f   = pos / FRAME;
    const int rem = pos % FRAME;
    const int hh  = rem / 30, ww = rem % 30;

    for (int idx = t; idx < NH * 64; idx += 256) {
        const int h = idx >> 6, i = idx & 63;
        const float ang = (i < 22) ? ff[f * 22 + i]
                        : (i < 43) ? fh[hh * 21 + (i - 22)]
                                   : fw[ww * 21 + (i - 43)];
        const float c = __cosf(ang), s = __sinf(ang);
        const int col = h * HD + 2 * i;
        const float qr = bf2f(qrow[col])     * rq * gq[col];
        const float qi = bf2f(qrow[col + 1]) * rq * gq[col + 1];
        const float kr = bf2f(krow[col])     * rk * gk[col];
        const float ki = bf2f(krow[col + 1]) * rk * gk[col + 1];
        const size_t base = ((size_t)h * L_TOK + pos) * HD + 2 * i;
        qb[base]     = f2bf(qr * c - qi * s);
        qb[base + 1] = f2bf(qr * s + qi * c);
        kb[base]     = f2bf(kr * c - ki * s);
        kb[base + 1] = f2bf(kr * s + ki * c);
    }
    for (int idx = t; idx < DIM; idx += 256) {
        const int h = idx >> 7, d = idx & 127;
        vT[((size_t)h * HD + d) * L_TOK + pos] = vrow[idx];
    }
}

// ---------------------------------------------------------------------------
// Flash attention, frame-causal. Grid (L/128, NH), 8 waves/block.
// Each wave owns a 16-row q strip; frame boundaries (480) are 16-aligned so
// causality becomes a k-loop bound -> no element masking.
// ---------------------------------------------------------------------------
__global__ __launch_bounds__(256) void attn_kernel(
    const unsigned short* __restrict__ qb,
    const unsigned short* __restrict__ kbuf,
    const unsigned short* __restrict__ vT,
    unsigned short* __restrict__ ctx)
{
    __shared__ unsigned short sP[8][16 * 64]; // per-wave P transpose staging

    const int lane = threadIdx.x & 31;
    const int wave = threadIdx.x >> 5;
    const int lr16 = lane & 15;
    const int hl   = lane >> 4;
    const int kbse = hl * 8;
    const int h    = blockIdx.y;
    const int qr0  = blockIdx.x * 128 + wave * 16;
    const float scale = 0.08838834764831845f; // 1/sqrt(128)

    const unsigned short* qh = qb   + (size_t)h * L_TOK * HD;
    const unsigned short* kh = kbuf + (size_t)h * L_TOK * HD;
    const unsigned short* vh = vT   + (size_t)h * HD * L_TOK;

    // Q strip fragments (4 chunks of K=32 over HD=128), kept in registers
    Frag16 aq[4];
#pragma unroll
    for (int c = 0; c < 4; c++) {
        const unsigned short* p = qh + (size_t)(qr0 + lr16) * HD + kbse + 32 * c;
        aq[c].u[0] = *(const uint4*)p;
        aq[c].u[1] = *(const uint4*)(p + 16);
    }

    float m[8], lsum[8];
    v8f acc[8];
#pragma unroll
    for (int r = 0; r < 8; r++) { m[r] = -1e30f; lsum[r] = 0.f; }
#pragma unroll
    for (int ni = 0; ni < 8; ni++) acc[ni] = vzero();

    const int kend = (qr0 / FRAME + 1) * FRAME;
    unsigned short* myP = sP[wave];

    for (int kt = 0; kt < kend; kt += 64) {
        // ---- S = Q K^T for a 16x64 score strip (up to 16 WMMAs) ----
        v8f s[4];
        bool vj[4];
#pragma unroll
        for (int j = 0; j < 4; j++) {
            const int kk0 = kt + 16 * j;
            vj[j] = (kk0 < kend);
            s[j] = vzero();
            if (vj[j]) {
#pragma unroll
                for (int c = 0; c < 4; c++) {
                    Frag16 fk;
                    const unsigned short* p =
                        kh + (size_t)(kk0 + lr16) * HD + kbse + 32 * c;
                    fk.u[0] = *(const uint4*)p;
                    fk.u[1] = *(const uint4*)(p + 16);
                    s[j] = wmma_bf16(aq[c].v, fk.v, s[j]);
                }
#pragma unroll
                for (int r = 0; r < 8; r++) s[j][r] *= scale;
            }
        }

        // ---- online softmax stats (per row; halves independent) ----
        float nm[8], alpha[8], rsum[8];
#pragma unroll
        for (int r = 0; r < 8; r++) {
            float v = -1e30f;
#pragma unroll
            for (int j = 0; j < 4; j++) if (vj[j]) v = fmaxf(v, s[j][r]);
#pragma unroll
            for (int o = 8; o >= 1; o >>= 1) v = fmaxf(v, __shfl_xor(v, o, 16));
            nm[r]    = fmaxf(m[r], v);
            alpha[r] = __expf(m[r] - nm[r]);
            m[r]     = nm[r];
            rsum[r]  = 0.f;
        }
        // P = exp(S - m) -> wave-private LDS (C-layout -> row-major 16x64)
#pragma unroll
        for (int j = 0; j < 4; j++) {
#pragma unroll
            for (int r = 0; r < 8; r++) {
                const float p = vj[j] ? __expf(s[j][r] - nm[r]) : 0.f;
                rsum[r] += p;
                myP[(r + 8 * hl) * 64 + 16 * j + lr16] = f2bf(p);
            }
        }
#pragma unroll
        for (int r = 0; r < 8; r++) {
            float v = rsum[r];
#pragma unroll
            for (int o = 8; o >= 1; o >>= 1) v += __shfl_xor(v, o, 16);
            lsum[r] = lsum[r] * alpha[r] + v;
        }
#pragma unroll
        for (int ni = 0; ni < 8; ni++)
#pragma unroll
            for (int r = 0; r < 8; r++) acc[ni][r] *= alpha[r];

        // ---- reload P in A-layout, then acc += P V (16 WMMAs) ----
        Frag16 ap[2];
#pragma unroll
        for (int c = 0; c < 2; c++) {
            const unsigned short* p = myP + lr16 * 64 + kbse + 32 * c;
            ap[c].u[0] = *(const uint4*)p;
            ap[c].u[1] = *(const uint4*)(p + 16);
        }
#pragma unroll
        for (int ni = 0; ni < 8; ni++) {
#pragma unroll
            for (int c = 0; c < 2; c++) {
                Frag16 fv;
                const unsigned short* p =
                    vh + (size_t)(16 * ni + lr16) * L_TOK + kt + kbse + 32 * c;
                fv.u[0] = *(const uint4*)p;
                fv.u[1] = *(const uint4*)(p + 16);
                acc[ni] = wmma_bf16(ap[c].v, fv.v, acc[ni]);
            }
        }
    }

    // ---- normalize and write context [L][DIM] (bf16) ----
#pragma unroll
    for (int ni = 0; ni < 8; ni++) {
#pragma unroll
        for (int r = 0; r < 8; r++) {
            const int row = qr0 + r + 8 * hl;
            const int col = h * HD + 16 * ni + lr16;
            const float o = acc[ni][r] / lsum[r];
            ctx[(size_t)row * DIM + col] = f2bf(o);
        }
    }
}

// ---------------------------------------------------------------------------
// Host orchestration
// ---------------------------------------------------------------------------
extern "C" void kernel_launch(void* const* d_in, const int* in_sizes, int n_in,
                              void* d_out, int out_size, void* d_ws, size_t ws_size,
                              hipStream_t stream)
{
    const float* x  = (const float*)d_in[0];
    const float* Wq = (const float*)d_in[1];
    const float* bq = (const float*)d_in[2];
    const float* Wk = (const float*)d_in[3];
    const float* bk = (const float*)d_in[4];
    const float* Wv = (const float*)d_in[5];
    const float* bv = (const float*)d_in[6];
    const float* Wo = (const float*)d_in[7];
    const float* bo = (const float*)d_in[8];
    const float* gq = (const float*)d_in[9];
    const float* gk = (const float*)d_in[10];
    const float* ff = (const float*)d_in[11];
    const float* fh = (const float*)d_in[12];
    const float* fw = (const float*)d_in[13];
    float* out = (float*)d_out;

    char* ws = (char*)d_ws;
    size_t off = 0;
    auto alloc = [&](size_t bytes) -> char* {
        char* p = ws + off;
        off += (bytes + 255) & ~(size_t)255;
        return p;
    };
    const size_t actB = (size_t)L_TOK * DIM * sizeof(unsigned short);
    const size_t wB   = (size_t)DIM * DIM * sizeof(unsigned short);

    unsigned short* xb  = (unsigned short*)alloc(actB);
    unsigned short* wqt = (unsigned short*)alloc(wB);
    unsigned short* wkt = (unsigned short*)alloc(wB);
    unsigned short* wvt = (unsigned short*)alloc(wB);
    unsigned short* wot = (unsigned short*)alloc(wB);
    unsigned short* Qg  = (unsigned short*)alloc(actB);
    unsigned short* Kg  = (unsigned short*)alloc(actB);
    unsigned short* Vg  = (unsigned short*)alloc(actB);
    unsigned short* qbf = (unsigned short*)alloc(actB);
    unsigned short* kbf = (unsigned short*)alloc(actB);
    unsigned short* vT  = (unsigned short*)alloc(actB);
    unsigned short* ctx = (unsigned short*)alloc(actB);
    (void)ws_size; (void)in_sizes; (void)n_in; (void)out_size;

    // 1. casts
    const int nx = L_TOK * DIM;
    cast_f32_bf16<<<(nx / 4 + 255) / 256, 256, 0, stream>>>(x, xb, nx);
    dim3 tg(DIM / 32, DIM / 32);
    cast_transpose<<<tg, 256, 0, stream>>>(Wq, wqt, DIM);
    cast_transpose<<<tg, 256, 0, stream>>>(Wk, wkt, DIM);
    cast_transpose<<<tg, 256, 0, stream>>>(Wv, wvt, DIM);
    cast_transpose<<<tg, 256, 0, stream>>>(Wo, wot, DIM);

    // 2. QKV projections (bf16 out, bias fused)
    dim3 gg(DIM / 128, L_TOK / 128);
    gemm_bf16<true><<<gg, 256, 0, stream>>>(xb, wqt, bq, Qg, L_TOK, DIM, DIM);
    gemm_bf16<true><<<gg, 256, 0, stream>>>(xb, wkt, bk, Kg, L_TOK, DIM, DIM);
    gemm_bf16<true><<<gg, 256, 0, stream>>>(xb, wvt, bv, Vg, L_TOK, DIM, DIM);

    // 3. RMSNorm + RoPE + layouts
    norm_rope_kernel<<<L_TOK, 256, 0, stream>>>(Qg, Kg, Vg, gq, gk, ff, fh, fw,
                                                qbf, kbf, vT);

    // 4. frame-causal flash attention
    dim3 ag(L_TOK / 128, NH);
    attn_kernel<<<ag, 256, 0, stream>>>(qbf, kbf, vT, ctx);

    // 5. output projection (f32 out)
    gemm_bf16<false><<<gg, 256, 0, stream>>>(ctx, wot, bo, out, L_TOK, DIM, DIM);
}
